// Attention_68719476736517
// MI455X (gfx1250) — matrix operand order repaired and checked by reading, not compiled, not run on hardware
//
#include <hip/hip_runtime.h>
#include <math.h>

#define B 64
#define T 256
#define H 768

typedef _Float16 h8  __attribute__((ext_vector_type(8)));
typedef _Float16 h16 __attribute__((ext_vector_type(16)));
typedef float    f8  __attribute__((ext_vector_type(8)));
typedef unsigned int u32x4 __attribute__((ext_vector_type(4)));
typedef int          i32x4 __attribute__((ext_vector_type(4)));
typedef int          i32x8 __attribute__((ext_vector_type(8)));

// WMMA f16 A-fragment as loaded here (16x32, same k order as B): per lane one contiguous 16-half run
// at K = laneH*16  (laneH = lane>>4). Row M = lane&15.
static __device__ __forceinline__ h16 ld_a16(const _Float16* p) {
  h8 lo = *(const h8*)(p);
  h8 hi = *(const h8*)(p + 8);
  return __builtin_shufflevector(lo, hi, 0,1,2,3,4,5,6,7,8,9,10,11,12,13,14,15);
}

static __device__ __forceinline__ f8 wmma_f16(h16 a, h16 b, f8 c) {
  return __builtin_amdgcn_wmma_f32_16x16x32_f16(false, a, false, b, (short)0, c,
                                                false, false);
}

static __device__ __forceinline__ f8 zero8() {
  f8 z = {0.f,0.f,0.f,0.f,0.f,0.f,0.f,0.f};
  return z;
}

// ---------------------------------------------------------------------------
// Prep kernels: build f16 (transposed where needed) operand copies in d_ws
// ---------------------------------------------------------------------------

__global__ void __launch_bounds__(256) prep_weights(
    const float* __restrict__ Wc1, const float* __restrict__ Wc2,
    const float* __restrict__ Wb,
    _Float16* __restrict__ wc1t, _Float16* __restrict__ wc2t,
    _Float16* __restrict__ wbT) {
  int idx = blockIdx.x * 256 + threadIdx.x;
  if (idx < H * H) {
    int n = idx / H, k = idx - n * H;
    wbT[idx] = (_Float16)Wb[k * H + n];
  }
  if (idx < T * H) {
    int i = idx / H, h = idx - i * H;
    wc1t[idx] = (_Float16)Wc1[h * T + i];
    wc2t[idx] = (_Float16)Wc2[h * T + i];
  }
}

__global__ void __launch_bounds__(256) prep_qp(
    const float* __restrict__ q, const float* __restrict__ p,
    const float* __restrict__ Wd,
    _Float16* __restrict__ q16, _Float16* __restrict__ p16,
    _Float16* __restrict__ pd16) {
  size_t idx = (size_t)blockIdx.x * 256 + threadIdx.x;
  int h = (int)(idx % H);
  float qv = q[idx], pv = p[idx];
  q16[idx]  = (_Float16)qv;
  p16[idx]  = (_Float16)pv;
  pd16[idx] = (_Float16)(pv * Wd[h]);
}

__global__ void __launch_bounds__(256) transpose_q(
    const float* __restrict__ q, _Float16* __restrict__ qT16) {
  size_t idx = (size_t)blockIdx.x * 256 + threadIdx.x;
  int b = (int)(idx / ((size_t)H * T));
  int rem = (int)(idx - (size_t)b * H * T);
  int h = rem / T;
  int t = rem - h * T;
  qT16[idx] = (_Float16)q[((size_t)b * T + t) * H + h];
}

__global__ void __launch_bounds__(256) rowdots(
    const float* __restrict__ q, const float* __restrict__ p,
    const float* __restrict__ Wm,
    float* __restrict__ qWm, float* __restrict__ pWm) {
  int wave = threadIdx.x >> 5, lane = threadIdx.x & 31;
  int row = blockIdx.x * 8 + wave;
  const float* qr = q + (size_t)row * H;
  const float* pr = p + (size_t)row * H;
  float aq = 0.f, ap = 0.f;
  for (int k = lane; k < H; k += 32) {
    float w = Wm[k];
    aq += qr[k] * w;
    ap += pr[k] * w;
  }
  #pragma unroll
  for (int off = 1; off < 32; off <<= 1) {
    aq += __shfl_xor(aq, off, 32);
    ap += __shfl_xor(ap, off, 32);
  }
  if (lane == 0) { qWm[row] = aq; pWm[row] = ap; }
}

// ---------------------------------------------------------------------------
// pb16 = f16(p @ Wb) — WMMA GEMM, M=B*T, N=H, K=H.
// The 64-row Wb^T B-strip is identical for all 8 waves, so it is staged into
// LDS by the Tensor Data Mover (one DMA per WG, double-buffered, TENSORcnt).
// ---------------------------------------------------------------------------
__global__ void __launch_bounds__(256) gemm_pb(
    const _Float16* __restrict__ p16, const _Float16* __restrict__ wbT,
    _Float16* __restrict__ pb16) {
  __shared__ alignas(128) _Float16 sB[2][64][32];   // 2 x (64 rows x 32 halves)

  const int mb = blockIdx.x / 12, nb = blockIdx.x % 12;
  const int wave = threadIdx.x >> 5, lane = threadIdx.x & 31;
  const int laneN = lane & 15, laneH = lane >> 4;
  const int m0 = mb * 128 + wave * 16;
  const int h0 = nb * 64;

  // LDS byte offsets of the two staging buffers (flat shared-aperture pointers
  // keep the LDS-space offset in the low 32 bits — ISA 10.2 aperture mapping).
  const unsigned lds0 = (unsigned)(uintptr_t)&sB[0][0][0];
  const unsigned lds1 = (unsigned)(uintptr_t)&sB[1][0][0];
  const unsigned long long gbase =
      (unsigned long long)(uintptr_t)(wbT + (size_t)h0 * H);

  // TDM D# (ISA ch.8): 2-D tile, data_size=2B, tile 32(dim0) x 64(dim1),
  // tensor_dim0 = stride = 768 elements; groups 2/3 unused (zero).
  auto tdm_issue = [&](int kk) {
    unsigned long long ga = gbase + (unsigned long long)(kk * 32) * 2u;
    unsigned lds = (kk & 1) ? lds1 : lds0;
    u32x4 g0 = { 1u,                                   // count=1 valid D#
                 lds,                                  // lds_addr (bytes)
                 (unsigned)ga,                         // global_addr[31:0]
                 ((unsigned)(ga >> 32) & 0x01FFFFFFu)  // global_addr[56:32]
                     | 0x80000000u };                  // type=2 ("image")
    i32x8 g1 = { 0x10000,                              // data_size=1 (2 bytes)
                 0x03000000,                           // tensor_dim0=768 (lo16 @48)
                 0x03000000,                           // tensor_dim1=768 (lo16 @80)
                 0x200000,                             // tile_dim0=32 (@112)
                 64,                                   // tile_dim1=64 (@128)
                 768,                                  // tensor_dim0_stride lo32
                 0, 0 };
    i32x4 gz4 = { 0, 0, 0, 0 };
    i32x8 gz8 = { 0, 0, 0, 0, 0, 0, 0, 0 };
    __builtin_amdgcn_tensor_load_to_lds(g0, g1, gz4, gz4, gz8, 0);
  };

  if (threadIdx.x < 32) tdm_issue(0);   // wave 0 owns the DMA pipeline

  f8 acc[4];
  #pragma unroll
  for (int nt = 0; nt < 4; ++nt) acc[nt] = zero8();

  const _Float16* pA = p16 + (size_t)(m0 + laneN) * H + laneH * 16;
  h16 aCur = ld_a16(pA);                // A fragment for chunk 0 (per-wave)

  for (int kk = 0; kk < 24; ++kk) {
    __syncthreads();                    // buf[(kk+1)&1] consumers (iter kk-1) done
    if (threadIdx.x < 32) {
      if (kk + 1 < 24) {
        tdm_issue(kk + 1);                       // next chunk in flight
        __builtin_amdgcn_s_wait_tensorcnt(1);    // chunk kk complete
      } else {
        __builtin_amdgcn_s_wait_tensorcnt(0);    // last chunk complete
      }
    }
    __syncthreads();                    // chunk kk now visible in LDS

    h16 aNext = aCur;
    if (kk + 1 < 24) aNext = ld_a16(pA + (kk + 1) * 32);  // prefetch next A

    const _Float16* bb = &sB[kk & 1][laneN][laneH * 16];
    acc[0] = wmma_f16(aCur, *(const h16*)(bb),            acc[0]);
    acc[1] = wmma_f16(aCur, *(const h16*)(bb + 16 * 32),  acc[1]);
    acc[2] = wmma_f16(aCur, *(const h16*)(bb + 32 * 32),  acc[2]);
    acc[3] = wmma_f16(aCur, *(const h16*)(bb + 48 * 32),  acc[3]);
    aCur = aNext;
  }

  #pragma unroll
  for (int nt = 0; nt < 4; ++nt)
    #pragma unroll
    for (int r = 0; r < 8; ++r)
      pb16[(size_t)(m0 + r + 8 * laneH) * H + h0 + nt * 16 + laneN] =
          (_Float16)acc[nt][r];
}

// ---------------------------------------------------------------------------
// Main fused attention kernel: one WG per (batch, 16-row i-tile).
// Score types: 0 = S1 (concat), 1 = S2 (bilinear), 2 = S3 (elem-prod),
//              3 = S4 (elem-diff)  -> matches (qc,qb,qd,qm) output order.
// ---------------------------------------------------------------------------
__global__ void __launch_bounds__(256) attn_main(
    const _Float16* __restrict__ q16,  const _Float16* __restrict__ p16,
    const _Float16* __restrict__ qT16, const _Float16* __restrict__ pd16,
    const _Float16* __restrict__ pb16, const _Float16* __restrict__ wc1t,
    const _Float16* __restrict__ wc2t, const float* __restrict__ qWm,
    const float* __restrict__ pWm,     const float* __restrict__ vc,
    const float* __restrict__ vd,      const float* __restrict__ vm,
    float* __restrict__ out) {
  __shared__ alignas(16) _Float16 sP[4][16][T];   // softmaxed probs (32 KB)
  __shared__ float red[4][16][8];                 // per-wave row partials
  __shared__ float grow[4][16];                   // row max, then row sum

  const int b    = blockIdx.x >> 4;
  const int i0   = (blockIdx.x & 15) << 4;
  const int wave = threadIdx.x >> 5;
  const int lane = threadIdx.x & 31;
  const int laneN = lane & 15;
  const int laneH = lane >> 4;

  // warm L2/WGP$ for the output-phase q^T rows this wave will consume later
  #pragma unroll
  for (int nt = 0; nt < 6; ++nt) {
    const int hcol = wave * 96 + nt * 16 + laneN;
    __builtin_prefetch(qT16 + ((size_t)b * H + hcol) * T + laneH * 256, 0, 1);
  }

  // ---- score GEMMs: each wave owns j-columns [wave*32, wave*32+32) ----
  f8 acc[4][2];
  #pragma unroll
  for (int t = 0; t < 4; ++t) { acc[t][0] = zero8(); acc[t][1] = zero8(); }

  const int rowA = i0 + laneN;
  const _Float16* pA_p  = p16  + (size_t)(b * T + rowA) * H + laneH * 16;
  const _Float16* pA_pb = pb16 + (size_t)(b * T + rowA) * H + laneH * 16;
  const _Float16* pA_pd = pd16 + (size_t)(b * T + rowA) * H + laneH * 16;
  const _Float16* pA_w  = wc1t + (size_t)rowA * H + laneH * 16;
  const int j0 = wave * 32;
  const _Float16* pBq0 = q16  + (size_t)(b * T + j0 + laneN) * H + laneH * 16;
  const _Float16* pBq1 = q16  + (size_t)(b * T + j0 + 16 + laneN) * H + laneH * 16;
  const _Float16* pBw0 = wc2t + (size_t)(j0 + laneN) * H + laneH * 16;
  const _Float16* pBw1 = wc2t + (size_t)(j0 + 16 + laneN) * H + laneH * 16;

  struct SF { h16 aP, aPB, aPD, aW, bq0, bw0, bq1, bw1; };
  auto loadS = [&](int k) -> SF {
    SF f;
    f.aP  = ld_a16(pA_p  + k);
    f.aPB = ld_a16(pA_pb + k);
    f.aPD = ld_a16(pA_pd + k);
    f.aW  = ld_a16(pA_w  + k);
    f.bq0 = *(const h16*)(pBq0 + k);
    f.bw0 = *(const h16*)(pBw0 + k);
    f.bq1 = *(const h16*)(pBq1 + k);
    f.bw1 = *(const h16*)(pBw1 + k);
    return f;
  };
  auto mmaS = [&](const SF& f) {
    acc[0][0] = wmma_f16(f.aP,  f.bw0, acc[0][0]);   // whp
    acc[1][0] = wmma_f16(f.aPB, f.bq0, acc[1][0]);   // (pWb) q^T
    acc[2][0] = wmma_f16(f.aPD, f.bq0, acc[2][0]);   // (p*Wd) q^T
    acc[3][0] = wmma_f16(f.aW,  f.bq0, acc[3][0]);   // whq^T
    acc[0][1] = wmma_f16(f.aP,  f.bw1, acc[0][1]);
    acc[1][1] = wmma_f16(f.aPB, f.bq1, acc[1][1]);
    acc[2][1] = wmma_f16(f.aPD, f.bq1, acc[2][1]);
    acc[3][1] = wmma_f16(f.aW,  f.bq1, acc[3][1]);
  };

  {
    SF cur = loadS(0);
    #pragma unroll 2
    for (int k0 = 32; k0 < H; k0 += 32) {
      SF nxt = loadS(k0);   // next-step fragments fly under current WMMAs
      mmaS(cur);
      cur = nxt;
    }
    mmaS(cur);
  }

  // ---- elementwise: C layout is M = r + 8*laneH, N = laneN ----
  #pragma unroll
  for (int n = 0; n < 2; ++n) {
    const int j = j0 + n * 16 + laneN;
    const float vdj = vd[j], vmj = vm[j], qwj = qWm[b * T + j];
    #pragma unroll
    for (int r = 0; r < 8; ++r) {
      const int i = i0 + r + 8 * laneH;
      float s1 = tanhf(acc[0][n][r] + acc[3][n][r]) * vc[i];  // concat attn
      float s4 = tanhf(qwj - pWm[b * T + i]) * vmj;           // diff attn
      acc[0][n][r] = s1;
      acc[2][n][r] = tanhf(acc[2][n][r]) * vdj;               // prod attn
      acc[3][n][r] = s4;                                      // s2 unchanged
    }
  }

  // ---- softmax over j (cross-wave): max ----
  #pragma unroll
  for (int t = 0; t < 4; ++t)
    #pragma unroll
    for (int r = 0; r < 8; ++r) {
      float m = fmaxf(acc[t][0][r], acc[t][1][r]);
      #pragma unroll
      for (int off = 1; off < 16; off <<= 1) m = fmaxf(m, __shfl_xor(m, off, 16));
      if (laneN == 0) red[t][r + 8 * laneH][wave] = m;
    }
  __syncthreads();
  if (threadIdx.x < 64) {
    int t = threadIdx.x >> 4, row = threadIdx.x & 15;
    float g = red[t][row][0];
    #pragma unroll
    for (int w = 1; w < 8; ++w) g = fmaxf(g, red[t][row][w]);
    grow[t][row] = g;
  }
  __syncthreads();

  // ---- exp + sum ----
  #pragma unroll
  for (int t = 0; t < 4; ++t)
    #pragma unroll
    for (int r = 0; r < 8; ++r) {
      const float g = grow[t][r + 8 * laneH];
      float e0 = __expf(acc[t][0][r] - g);
      float e1 = __expf(acc[t][1][r] - g);
      acc[t][0][r] = e0; acc[t][1][r] = e1;
      float s = e0 + e1;
      #pragma unroll
      for (int off = 1; off < 16; off <<= 1) s += __shfl_xor(s, off, 16);
      if (laneN == 0) red[t][r + 8 * laneH][wave] = s;
    }
  __syncthreads();
  if (threadIdx.x < 64) {
    int t = threadIdx.x >> 4, row = threadIdx.x & 15;
    float s = 0.f;
    #pragma unroll
    for (int w = 0; w < 8; ++w) s += red[t][row][w];
    grow[t][row] = s;
  }
  __syncthreads();

  // ---- normalized probs -> LDS (f16) ----
  #pragma unroll
  for (int t = 0; t < 4; ++t)
    #pragma unroll
    for (int r = 0; r < 8; ++r) {
      const int row = r + 8 * laneH;
      const float rinv = 1.f / grow[t][row];
      sP[t][row][j0 + laneN]      = (_Float16)(acc[t][0][r] * rinv);
      sP[t][row][j0 + 16 + laneN] = (_Float16)(acc[t][1][r] * rinv);
    }
  __syncthreads();

  // ---- outputs: out_t[i, h] = sum_j P_t[i,j] * q[b,j,h]; wave owns 96 h-cols
  const size_t outStride = (size_t)B * T * H;
  for (int nt = 0; nt < 6; ++nt) {
    const int hcol = wave * 96 + nt * 16 + laneN;
    const _Float16* pB = qT16 + ((size_t)b * H + hcol) * T + laneH * 16;
    f8 o[4];
    #pragma unroll
    for (int t = 0; t < 4; ++t) o[t] = zero8();

    struct OF { h16 bq, ap0, ap1, ap2, ap3; };
    auto loadO = [&](int k) -> OF {
      OF f;
      f.bq  = *(const h16*)(pB + k);
      f.ap0 = ld_a16(&sP[0][laneN][k + laneH * 16]);
      f.ap1 = ld_a16(&sP[1][laneN][k + laneH * 16]);
      f.ap2 = ld_a16(&sP[2][laneN][k + laneH * 16]);
      f.ap3 = ld_a16(&sP[3][laneN][k + laneH * 16]);
      return f;
    };
    auto mmaO = [&](const OF& f) {
      o[0] = wmma_f16(f.ap0, f.bq, o[0]);
      o[1] = wmma_f16(f.ap1, f.bq, o[1]);
      o[2] = wmma_f16(f.ap2, f.bq, o[2]);
      o[3] = wmma_f16(f.ap3, f.bq, o[3]);
    };

    OF cur = loadO(0);
    #pragma unroll 2
    for (int k0 = 32; k0 < T; k0 += 32) {
      OF nxt = loadO(k0);
      mmaO(cur);
      cur = nxt;
    }
    mmaO(cur);

    #pragma unroll
    for (int t = 0; t < 4; ++t)
      #pragma unroll
      for (int r = 0; r < 8; ++r) {
        const int i = i0 + r + 8 * laneH;
        out[(size_t)t * outStride + ((size_t)b * T + i) * H + hcol] = o[t][r];
      }
  }
}

// ---------------------------------------------------------------------------
extern "C" void kernel_launch(void* const* d_in, const int* in_sizes, int n_in,
                              void* d_out, int out_size, void* d_ws,
                              size_t ws_size, hipStream_t stream) {
  const float* q   = (const float*)d_in[0];
  const float* p   = (const float*)d_in[1];
  const float* Wc1 = (const float*)d_in[2];
  const float* Wc2 = (const float*)d_in[3];
  const float* vc  = (const float*)d_in[4];
  const float* Wb  = (const float*)d_in[5];
  const float* Wd  = (const float*)d_in[6];
  const float* vd  = (const float*)d_in[7];
  const float* Wm  = (const float*)d_in[8];
  const float* vm  = (const float*)d_in[9];
  float* out = (float*)d_out;

  // workspace layout (all offsets 256B aligned)
  const size_t SZ_BTH16 = (size_t)B * T * H * sizeof(_Float16);  // 25,165,824
  char* ws = (char*)d_ws;
  _Float16* q16  = (_Float16*)(ws);
  _Float16* p16  = (_Float16*)(ws + SZ_BTH16);
  _Float16* qT16 = (_Float16*)(ws + 2 * SZ_BTH16);
  _Float16* pd16 = (_Float16*)(ws + 3 * SZ_BTH16);
  _Float16* pb16 = (_Float16*)(ws + 4 * SZ_BTH16);
  char* wp = ws + 5 * SZ_BTH16;
  _Float16* wc1t = (_Float16*)(wp);                      // T*H*2 = 393,216
  _Float16* wc2t = (_Float16*)(wp + 393216);
  _Float16* wbT  = (_Float16*)(wp + 2 * 393216);         // H*H*2 = 1,179,648
  float* qWm = (float*)(wp + 2 * 393216 + 1179648);      // B*T*4 = 65,536
  float* pWm = (float*)(wp + 2 * 393216 + 1179648 + 65536);

  const int NBTH = B * T * H;  // 12,582,912

  prep_weights<<<(H * H) / 256, 256, 0, stream>>>(Wc1, Wc2, Wb, wc1t, wc2t, wbT);
  prep_qp<<<NBTH / 256, 256, 0, stream>>>(q, p, Wd, q16, p16, pd16);
  transpose_q<<<NBTH / 256, 256, 0, stream>>>(q, qT16);
  rowdots<<<(B * T) / 8, 256, 0, stream>>>(q, p, Wm, qWm, pWm);
  gemm_pb<<<((B * T) / 128) * (H / 64), 256, 0, stream>>>(p16, wbT, pb16);
  attn_main<<<B * (T / 16), 256, 0, stream>>>(q16, p16, qT16, pd16, pb16, wc1t,
                                              wc2t, qWm, pWm, vc, vd, vm, out);
}
